// GatedMoE_62732292325615
// MI455X (gfx1250) — matrix operand order repaired and checked by reading, not compiled, not run on hardware
//
#include <hip/hip_runtime.h>
#include <hip/hip_bf16.h>

typedef __attribute__((ext_vector_type(16))) __bf16 v16bf;
typedef __attribute__((ext_vector_type(8)))  float  v8f;

union FragBF {
    v16bf    v;
    unsigned u[8];
};

__device__ __forceinline__ unsigned short f2bf(float f) {
    unsigned u = __float_as_uint(f);
    u += 0x7FFFu + ((u >> 16) & 1u);          // round-to-nearest-even
    return (unsigned short)(u >> 16);
}

__device__ __forceinline__ unsigned pack_bf16(float lo, float hi) {
    unsigned a = __float_as_uint(lo);
    unsigned b = __float_as_uint(hi);
    a = (a + 0x7FFFu + ((a >> 16) & 1u)) >> 16;
    b = (b + 0x7FFFu + ((b >> 16) & 1u)) & 0xFFFF0000u;
    return a | b;
}

// Fragment stored swizzled as [frag][lane][8 dwords]: two b128 loads per lane.
__device__ __forceinline__ void load_frag(const unsigned* __restrict__ base, int lane, FragBF& f) {
    const uint4* p = (const uint4*)base + (size_t)lane * 2;
    uint4 q0 = p[0];
    uint4 q1 = p[1];
    f.u[0] = q0.x; f.u[1] = q0.y; f.u[2] = q0.z; f.u[3] = q0.w;
    f.u[4] = q1.x; f.u[5] = q1.y; f.u[6] = q1.z; f.u[7] = q1.w;
}

__device__ __forceinline__ v8f wmma_bf16(const FragBF& a, const FragBF& b, v8f c) {
    return __builtin_amdgcn_wmma_f32_16x16x32_bf16(false, a.v, false, b.v,
                                                   (short)0, c, false, false);
}

// ---------------------------------------------------------------- zero output
__global__ void zero_out(float* __restrict__ out, size_t n) {
    size_t i = (size_t)blockIdx.x * blockDim.x + threadIdx.x;
    if (i < n) out[i] = 0.0f;
}

// ---------------------------------------------------------------- x -> bf16 (pairs packed as dwords)
__global__ void cvt_x_bf16(const float* __restrict__ x, unsigned* __restrict__ xbf, size_t nPairs) {
    size_t i = (size_t)blockIdx.x * blockDim.x + threadIdx.x;
    if (i < nPairs) xbf[i] = pack_bf16(x[2 * i], x[2 * i + 1]);
}

// ---------------------------------------------------------------- W1 -> bf16, B-fragment swizzled
// Output dword id = ((((e*CT + ct)*KT + kt)*32 + lane)*8 + v)
__global__ void swz_w1(const float* __restrict__ W1, unsigned* __restrict__ w1s,
                       int D, int H, size_t total) {
    size_t id = (size_t)blockIdx.x * blockDim.x + threadIdx.x;
    if (id >= total) return;
    const int KT = D >> 5, CT = (2 * H) >> 4;
    int v    = (int)(id & 7);
    int lane = (int)((id >> 3) & 31);
    size_t rest = id >> 8;
    int kt = (int)(rest % KT); rest /= KT;
    int ct = (int)(rest % CT);
    int e  = (int)(rest / CT);
    int col = ct * 16 + (lane & 15);
    int d   = kt * 32 + ((lane & 16) >> 1) + ((v & 4) ? 16 : 0) + 2 * (v & 3);
    const size_t TWOH = 2 * (size_t)H;
    const float* p = W1 + ((size_t)e * D + d) * TWOH + col;
    w1s[id] = pack_bf16(p[0], p[TWOH]);
}

// ---------------------------------------------------------------- W2 -> bf16, B-fragment swizzled
__global__ void swz_w2(const float* __restrict__ W2, unsigned* __restrict__ w2s,
                       int D, int H, size_t total) {
    size_t id = (size_t)blockIdx.x * blockDim.x + threadIdx.x;
    if (id >= total) return;
    const int KT = H >> 5, CT = D >> 4;
    int v    = (int)(id & 7);
    int lane = (int)((id >> 3) & 31);
    size_t rest = id >> 8;
    int kt = (int)(rest % KT); rest /= KT;
    int ct = (int)(rest % CT);
    int e  = (int)(rest / CT);
    int col = ct * 16 + (lane & 15);
    int d   = kt * 32 + ((lane & 16) >> 1) + ((v & 4) ? 16 : 0) + 2 * (v & 3);
    const float* p = W2 + ((size_t)e * H + d) * D + col;
    w2s[id] = pack_bf16(p[0], p[D]);
}

// ---------------------------------------------------------------- router: logits, top-2, softmax
__global__ void moe_router(const float* __restrict__ x, const float* __restrict__ Wg,
                           int* __restrict__ idx, float* __restrict__ wgt,
                           int N, int D) {
    int t = blockIdx.x * blockDim.x + threadIdx.x;
    if (t >= N) return;
    float logits[8];
#pragma unroll
    for (int e = 0; e < 8; ++e) logits[e] = 0.0f;
    const float* xr = x + (size_t)t * D;
    for (int d = 0; d < D; ++d) {
        float xv = xr[d];
        const float* wr = Wg + (size_t)d * 8;
#pragma unroll
        for (int e = 0; e < 8; ++e) logits[e] = fmaf(xv, wr[e], logits[e]);
    }
    int i0 = 0; float s0 = logits[0];
#pragma unroll
    for (int e = 1; e < 8; ++e) if (logits[e] > s0) { s0 = logits[e]; i0 = e; }
    int i1 = -1; float s1 = -3.4e38f;
#pragma unroll
    for (int e = 0; e < 8; ++e) if (e != i0 && logits[e] > s1) { s1 = logits[e]; i1 = e; }
    float w0 = 1.0f / (1.0f + __expf(s1 - s0));
    idx[t * 2 + 0] = i0;  idx[t * 2 + 1] = i1;
    wgt[t * 2 + 0] = w0;  wgt[t * 2 + 1] = 1.0f - w0;
}

// ---------------------------------------------------------------- deterministic capacity assignment
__global__ void moe_assign(const int* __restrict__ idx, const float* __restrict__ wgt,
                           int* __restrict__ buf, float* __restrict__ wslot,
                           int N, int capacity, int capPad) {
    int th = threadIdx.x;
    if (th >= 16) return;
    int k = th >> 3, e = th & 7;
    int*   b = buf   + ((size_t)k * 8 + e) * capPad;
    float* w = wslot + ((size_t)k * 8 + e) * capPad;
    int pos = 0;
    for (int t = 0; t < N && pos < capacity; ++t) {
        if (idx[t * 2 + k] == e) { b[pos] = t; w[pos] = wgt[t * 2 + k]; ++pos; }
    }
    for (; pos < capPad; ++pos) { b[pos] = N; w[pos] = 0.0f; }
}

// ---------------------------------------------------------------- GEMM1 + gated SiLU
// grid: (capPad/128, H/32, E); block 256 = 8 waves.
// Wave macro-tile: 32 rows x 16 act cols -> 2 A frags x (b1,b2) = 4 WMMAs/K-step.
// Double-buffered LDS A-staging; writes act in GEMM2 A-fragment swizzle.
__global__ __launch_bounds__(256)
void moe_gemm1(const unsigned short* __restrict__ xbf, const unsigned* __restrict__ w1s,
               const int* __restrict__ bufK, unsigned short* __restrict__ actS,
               int N, int D, int H, int capPad) {
    const int e     = blockIdx.z;
    const int mBase = blockIdx.x * 128;
    const int tid  = threadIdx.x;
    const int lane = tid & 31;
    const int wid  = tid >> 5;
    const int rgrp = wid & 3;                       // 4 row groups of 32 rows
    const int csub = wid >> 2;                      // 2 col groups of 16 act cols
    const int KT1 = D >> 5;
    const int CT1 = (2 * H) >> 4;
    const int ct1 = blockIdx.y * 2 + csub;          // h1 column tile
    const int ct2 = ct1 + (H >> 4);                 // h2 column tile

    __shared__ int rows[128];
    __shared__ unsigned short atile[2][128 * 34];   // double buffer, padded stride

    if (tid < 128) rows[tid] = bufK[(size_t)e * capPad + mBase + tid];
    __syncthreads();

    const int m     = lane & 15;
    const int kHalf = (lane & 16) ? 8 : 0;
    const int sr = tid >> 1;                        // staging: 16 bf16 per thread
    const int sc = (tid & 1) * 16;

    // stage K-tile 0 into buffer 0
    {
        int t = rows[sr];
        uint4 q0 = {0u,0u,0u,0u}, q1 = {0u,0u,0u,0u};
        if (t < N) {
            const uint4* src = (const uint4*)(xbf + (size_t)t * D + sc);
            q0 = src[0]; q1 = src[1];
        }
        unsigned* dst = (unsigned*)&atile[0][sr * 34 + sc];
        dst[0] = q0.x; dst[1] = q0.y; dst[2] = q0.z; dst[3] = q0.w;
        dst[4] = q1.x; dst[5] = q1.y; dst[6] = q1.z; dst[7] = q1.w;
    }

    v8f acc1[2] = {v8f{}, v8f{}};
    v8f acc2[2] = {v8f{}, v8f{}};

    for (int kt = 0; kt < KT1; ++kt) {
        __syncthreads();
        if (kt + 1 < KT1) {                         // stage next K-tile into other buffer
            int t = rows[sr];
            uint4 q0 = {0u,0u,0u,0u}, q1 = {0u,0u,0u,0u};
            if (t < N) {
                const uint4* src = (const uint4*)(xbf + (size_t)t * D + (kt + 1) * 32 + sc);
                q0 = src[0]; q1 = src[1];
            }
            unsigned* dst = (unsigned*)&atile[(kt + 1) & 1][sr * 34 + sc];
            dst[0] = q0.x; dst[1] = q0.y; dst[2] = q0.z; dst[3] = q0.w;
            dst[4] = q1.x; dst[5] = q1.y; dst[6] = q1.z; dst[7] = q1.w;
        }

        const unsigned short* cur = atile[kt & 1];
        FragBF a0, a1;
#pragma unroll
        for (int v = 0; v < 8; ++v) {               // A 16x32 bf16 lane layout
            int kb = kHalf + ((v & 4) ? 16 : 0) + 2 * (v & 3);
            a0.u[v] = *(const unsigned*)&cur[(rgrp * 32 + m) * 34 + kb];
            a1.u[v] = *(const unsigned*)&cur[(rgrp * 32 + 16 + m) * 34 + kb];
        }
        FragBF b1, b2;
        load_frag(w1s + (((size_t)e * CT1 + ct1) * KT1 + kt) * 256, lane, b1);
        load_frag(w1s + (((size_t)e * CT1 + ct2) * KT1 + kt) * 256, lane, b2);

        acc1[0] = wmma_bf16(a0, b1, acc1[0]);
        acc2[0] = wmma_bf16(a0, b2, acc2[0]);
        acc1[1] = wmma_bf16(a1, b1, acc1[1]);
        acc2[1] = wmma_bf16(a1, b2, acc2[1]);
    }

    // act = h1 * silu(h2), stored in GEMM2 A-fragment swizzle.
    const int RT  = capPad >> 4;
    const int KT2 = H >> 5;
    const int m0  = (lane & 16) ? 8 : 0;
    const int h   = blockIdx.y * 32 + csub * 16 + m;   // act column
    const int kt2 = h >> 5;
    const int kkk = h & 31;
    const int bsel = (kkk >> 3) & 1;
    const int vsel = ((kkk >> 4) << 2) + ((kkk & 7) >> 1);
    const int half = kkk & 1;
#pragma unroll
    for (int r = 0; r < 2; ++r) {
        const int rt = (mBase >> 4) + rgrp * 2 + r;
        const size_t fb = (((size_t)e * RT + rt) * KT2 + kt2) * 256;   // dwords
#pragma unroll
        for (int v = 0; v < 8; ++v) {
            int rr = v + m0;
            float h1 = acc1[r][v], h2 = acc2[r][v];
            float av = h1 * h2 * (1.0f / (1.0f + __expf(-h2)));
            actS[(fb + (size_t)(rr + 16 * bsel) * 8 + vsel) * 2 + half] = f2bf(av);
        }
    }
}

// ---------------------------------------------------------------- GEMM2 + weighted scatter
// grid: (capPad/128, D/64, E); block 256 = 8 waves.
// Wave macro-tile: 32 rows x 32 cols -> 2 A + 2 B frags = 4 WMMAs/K-step.
__global__ __launch_bounds__(256)
void moe_gemm2(const unsigned* __restrict__ actS, const unsigned* __restrict__ w2s,
               const int* __restrict__ bufK, const float* __restrict__ wslotK,
               float* __restrict__ out, int N, int D, int H, int capPad) {
    const int e     = blockIdx.z;
    const int mBase = blockIdx.x * 128;
    const int tid  = threadIdx.x;
    const int lane = tid & 31;
    const int wid  = tid >> 5;
    const int rgrp = wid & 3;                       // 4 row groups of 32 rows
    const int cgrp = wid >> 2;                      // 2 col groups of 32 cols
    const int RT = capPad >> 4;
    const int KT = H >> 5;
    const int CT = D >> 4;
    const int rt0 = (mBase >> 4) + rgrp * 2;
    const int ct0 = blockIdx.y * 4 + cgrp * 2;

    v8f acc[2][2] = {{v8f{}, v8f{}}, {v8f{}, v8f{}}};
    const unsigned* a0Base = actS + (((size_t)e * RT + rt0)     * KT) * 256;
    const unsigned* a1Base = actS + (((size_t)e * RT + rt0 + 1) * KT) * 256;
    const unsigned* b0Base = w2s  + (((size_t)e * CT + ct0)     * KT) * 256;
    const unsigned* b1Base = w2s  + (((size_t)e * CT + ct0 + 1) * KT) * 256;

    for (int kt = 0; kt < KT; ++kt) {
        FragBF a0, a1, b0, b1;
        load_frag(a0Base + (size_t)kt * 256, lane, a0);
        load_frag(a1Base + (size_t)kt * 256, lane, a1);
        load_frag(b0Base + (size_t)kt * 256, lane, b0);
        load_frag(b1Base + (size_t)kt * 256, lane, b1);
        acc[0][0] = wmma_bf16(a0, b0, acc[0][0]);
        acc[0][1] = wmma_bf16(a0, b1, acc[0][1]);
        acc[1][0] = wmma_bf16(a1, b0, acc[1][0]);
        acc[1][1] = wmma_bf16(a1, b1, acc[1][1]);
    }

    const int m0 = (lane & 16) ? 8 : 0;
#pragma unroll
    for (int r = 0; r < 2; ++r) {
#pragma unroll
        for (int c = 0; c < 2; ++c) {
            const int colBase = (ct0 + c) * 16 + (lane & 15);
#pragma unroll
            for (int v = 0; v < 8; ++v) {
                int slot = mBase + rgrp * 32 + r * 16 + v + m0;
                int t = bufK[(size_t)e * capPad + slot];
                if (t < N) {
                    float w = wslotK[(size_t)e * capPad + slot];
                    out[(size_t)t * D + colBase] += w * acc[r][c][v];
                }
            }
        }
    }
}

// ---------------------------------------------------------------- host side
extern "C" void kernel_launch(void* const* d_in, const int* in_sizes, int n_in,
                              void* d_out, int out_size, void* d_ws, size_t ws_size,
                              hipStream_t stream) {
    const float* x  = (const float*)d_in[0];
    const float* Wg = (const float*)d_in[1];
    const float* W1 = (const float*)d_in[2];
    const float* W2 = (const float*)d_in[3];
    float* out = (float*)d_out;

    const int E = 8;
    const int D = in_sizes[1] / E;                       // 768
    const int N = in_sizes[0] / D;                       // 4096
    const int H = in_sizes[3] / (E * D);                 // 2048
    const int cap    = (int)(((long long)N * 12) / (10 * E));   // floor(1.2*N/E) = 614
    const int capPad = ((cap + 127) / 128) * 128;               // 640

    char* ws = (char*)d_ws;
    size_t off = 0;
    int*   idx   = (int*)  (ws + off); off += (size_t)N * 2 * sizeof(int);
    float* wgt   = (float*)(ws + off); off += (size_t)N * 2 * sizeof(float);
    int*   buf   = (int*)  (ws + off); off += (size_t)2 * E * capPad * sizeof(int);
    float* wslot = (float*)(ws + off); off += (size_t)2 * E * capPad * sizeof(float);
    off = (off + 255) & ~(size_t)255;
    unsigned short* xbf = (unsigned short*)(ws + off); off += (size_t)N * D * 2;
    off = (off + 255) & ~(size_t)255;
    unsigned short* actS = (unsigned short*)(ws + off); off += (size_t)E * capPad * H * 2;
    off = (off + 255) & ~(size_t)255;
    unsigned* w1s = (unsigned*)(ws + off); off += (size_t)E * 2 * H * D * 2;   // bf16 bytes
    off = (off + 255) & ~(size_t)255;
    unsigned* w2s = (unsigned*)(ws + off); off += (size_t)E * H * D * 2;

    const size_t nOut   = (size_t)N * D;
    const size_t xPairs = (size_t)N * D / 2;
    const size_t w1tot  = (size_t)E * H * D;             // dwords (= E*2H*D bf16 / 2)
    const size_t w2tot  = (size_t)E * H * D / 2;         // dwords

    zero_out  <<<dim3((unsigned)((nOut   + 255) / 256)), 256, 0, stream>>>(out, nOut);
    cvt_x_bf16<<<dim3((unsigned)((xPairs + 255) / 256)), 256, 0, stream>>>(x, (unsigned*)xbf, xPairs);
    swz_w1    <<<dim3((unsigned)((w1tot  + 255) / 256)), 256, 0, stream>>>(W1, w1s, D, H, w1tot);
    swz_w2    <<<dim3((unsigned)((w2tot  + 255) / 256)), 256, 0, stream>>>(W2, w2s, D, H, w2tot);
    moe_router<<<dim3((N + 255) / 256), 256, 0, stream>>>(x, Wg, idx, wgt, N, D);
    moe_assign<<<1, 32, 0, stream>>>(idx, wgt, buf, wslot, N, cap, capPad);

    for (int k = 0; k < 2; ++k) {
        const int*   bufK = buf   + (size_t)k * E * capPad;
        const float* wsK  = wslot + (size_t)k * E * capPad;
        moe_gemm1<<<dim3(capPad / 128, H / 32, E), 256, 0, stream>>>(
            xbf, w1s, bufK, actS, N, D, H, capPad);
        moe_gemm2<<<dim3(capPad / 128, D / 64, E), 256, 0, stream>>>(
            (const unsigned*)actS, w2s, bufK, wsK, out, N, D, H, capPad);
    }
}
